// SparsePromptGenerator_29557964931403
// MI455X (gfx1250) — matrix operand order repaired and checked by reading, not compiled, not run on hardware
//
#include <hip/hip_runtime.h>
#include <math.h>

// ---- problem constants (from reference) ----
#define B_    16
#define C_    256
#define H_    128
#define W_    128
#define HW_   (H_*W_)
#define NQ_   72
#define NQP_  80     // padded to 5 WMMA m-tiles
#define NTOK_ 8
#define ADIM_ 64
#define HID_  512

#if defined(__HIP_DEVICE_COMPILE__) && \
    __has_builtin(__builtin_amdgcn_global_load_async_to_lds_b128) && \
    __has_builtin(__builtin_amdgcn_s_wait_asynccnt)
#define USE_ASYNC 1
#else
#define USE_ASYNC 0
#endif

#define AS1 __attribute__((address_space(1)))
#define AS3 __attribute__((address_space(3)))

typedef int v4i_vs __attribute__((vector_size(4 * sizeof(int))));

typedef __attribute__((ext_vector_type(16))) __bf16 bf16x16;
typedef __attribute__((ext_vector_type(8)))  float  f32x8;

union FragBF { bf16x16 v; uint4 q[2]; };

#if USE_ASYNC
// one async b128 (16B) global->LDS transfer per lane
__device__ __forceinline__ void async_b128(const __bf16* g, __bf16* l) {
  __builtin_amdgcn_global_load_async_to_lds_b128(
      (AS1 v4i_vs*)(v4i_vs*)(void*)g,
      (AS3 v4i_vs*)(v4i_vs*)(void*)l, 0, 0);
}
#endif

// copy 32B (2 x b128) global->LDS, async if available
__device__ __forceinline__ void cp32(const __bf16* g, __bf16* l) {
#if USE_ASYNC
  async_b128(g,     l);
  async_b128(g + 8, l + 8);
#else
  const uint4* s = (const uint4*)g;
  uint4* d = (uint4*)l;
  d[0] = s[0]; d[1] = s[1];
#endif
}

// copy 64B (4 x b128) global->LDS, async if available
__device__ __forceinline__ void cp64(const __bf16* g, __bf16* l) {
#if USE_ASYNC
  async_b128(g,      l);
  async_b128(g + 8,  l + 8);
  async_b128(g + 16, l + 16);
  async_b128(g + 24, l + 24);
#else
  const uint4* s = (const uint4*)g;
  uint4* d = (uint4*)l;
  d[0] = s[0]; d[1] = s[1]; d[2] = s[2]; d[3] = s[3];
#endif
}

// A fragment: 16x32 bf16, LDS layout [m][k], 32 contiguous bf16 per row (64B).
__device__ __forceinline__ bf16x16 frag_a(const __bf16* base) {
  int lane = threadIdx.x & 31;
  int m  = lane & 15;
  int kh = lane >> 4;
  const char* pc = (const char*)base + m*64 + kh*16;
  FragBF u;
  u.q[0] = *(const uint4*)(pc);
  u.q[1] = *(const uint4*)(pc + 32);
  return u.v;
}

// B fragment: 32x16 bf16, LDS layout [n][k], 32 contiguous bf16 per row (64B).
__device__ __forceinline__ bf16x16 frag_b(const __bf16* base) {
  int lane = threadIdx.x & 31;
  int n  = lane & 15;
  int kh = lane >> 4;
  const char* pc = (const char*)base + n*64 + kh*16;
  FragBF u;
  u.q[0] = *(const uint4*)(pc);
  u.q[1] = *(const uint4*)(pc + 32);
  return u.v;
}

__device__ __forceinline__ float gelu_f(float x) {
  return 0.5f * x * (1.0f + erff(x * 0.70710678118654752f));
}

// ---------------- weight prepack: Wc (co,ci,3,3) f32 -> Wp [tap][co][ci] bf16 ----------------
__global__ void __launch_bounds__(256) pack_weights_kernel(const float* __restrict__ Wc,
                                                           __bf16* __restrict__ Wp) {
  int idx = blockIdx.x * 256 + threadIdx.x;           // 9*256*256
  if (idx >= 9 * C_ * C_) return;
  int tap = idx % 9;
  int rem = idx / 9;
  int ci  = rem % C_;
  int co  = rem / C_;
  float v = Wc[(size_t)co * (C_*9) + ci*9 + tap];
  Wp[(size_t)tap * (C_*C_) + co * C_ + ci] = (__bf16)v;
}

// ---------------- dilated conv as implicit GEMM (bf16 WMMA, f32 accum) ----------------
// M=co tile 128, N=64 w-positions in one row h, K=2304 (72 chunks of 32)
// double-buffered LDS, async weight staging
__global__ void __launch_bounds__(256) conv_wmma_kernel(const float* __restrict__ in,
                                                        const __bf16* __restrict__ Wp,
                                                        const float* __restrict__ bc,
                                                        __bf16* __restrict__ img_bf,
                                                        __bf16* __restrict__ imgT_bf) {
  __shared__ __align__(16) __bf16 ldsA[2][128 * 32];
  __shared__ __align__(16) __bf16 ldsB[2][64 * 32];

  int wg     = blockIdx.x;
  int co_blk = wg & 1;
  int w_blk  = (wg >> 1) & 1;
  int h      = (wg >> 2) & 127;
  int bidx   = wg >> 9;
  int tid    = threadIdx.x;
  int lane   = tid & 31;
  int wid    = tid >> 5;
  int wave_m = wid & 3;    // 4 waves x 32 co
  int wave_n = wid >> 2;   // 2 waves x 32 w
  int w_base = w_blk * 64;

  const f32x8 zero8 = {0.f,0.f,0.f,0.f,0.f,0.f,0.f,0.f};
  f32x8 acc[2][2];
  acc[0][0] = zero8; acc[0][1] = zero8; acc[1][0] = zero8; acc[1][1] = zero8;

  int sa_m    = tid >> 1;       // A-stage: 2 threads per row, 32B each
  int sa_half = tid & 1;
  int sb_n    = tid & 63;       // B-stage: 4 thread-groups over k
  int sb_kq   = tid >> 6;

  const __bf16* wA = Wp + (co_blk * 128 + sa_m) * C_ + sa_half * 16;
  const float*  inb = in + ((size_t)bidx * C_) * HW_;

  // stage A (weights, bf16) for chunk kk into buffer buf  [async]
  auto issueA = [&](int kk, int buf) {
    int tap = kk >> 3, ci0 = (kk & 7) * 32;
    cp32(wA + (size_t)tap * (C_*C_) + ci0, &ldsA[buf][sa_m * 32 + sa_half * 16]);
  };
  // stage B (input, f32 -> bf16 transpose-convert) for chunk kk into buffer buf [manual]
  auto issueB = [&](int kk, int buf) {
    int tap = kk >> 3, ci0 = (kk & 7) * 32;
    int kh = tap / 3, kw = tap % 3;
    int hp = h + 2 * (kh - 1);
    int wp = w_base + sb_n + 2 * (kw - 1);
    bool wok = ((unsigned)hp < (unsigned)H_) && ((unsigned)wp < (unsigned)W_);
    const float* src = inb + (size_t)(ci0 + sb_kq * 8) * HW_ + (wok ? (hp * W_ + wp) : 0);
    #pragma unroll
    for (int j = 0; j < 8; ++j) {
      float v = wok ? src[(size_t)j * HW_] : 0.0f;
      ldsB[buf][sb_n * 32 + sb_kq * 8 + j] = (__bf16)v;
    }
  };

  issueA(0, 0);
  issueB(0, 0);
  for (int kk = 0; kk < 72; ++kk) {
    int cur = kk & 1;
    if (kk < 71) {
      issueA(kk + 1, cur ^ 1);
      issueB(kk + 1, cur ^ 1);
    }
#if USE_ASYNC
    if (kk < 71) __builtin_amdgcn_s_wait_asynccnt(2);  // this wave's 2 outstanding = next A stage
    else         __builtin_amdgcn_s_wait_asynccnt(0);
#endif
    __syncthreads();

    FragBF a0, a1, b0, b1;
    a0.v = frag_a(&ldsA[cur][(wave_m * 32)      * 32]);
    a1.v = frag_a(&ldsA[cur][(wave_m * 32 + 16) * 32]);
    b0.v = frag_b(&ldsB[cur][(wave_n * 32)      * 32]);
    b1.v = frag_b(&ldsB[cur][(wave_n * 32 + 16) * 32]);

    acc[0][0] = __builtin_amdgcn_wmma_f32_16x16x32_bf16(false, a0.v, false, b0.v, (short)0, acc[0][0], false, false);
    acc[0][1] = __builtin_amdgcn_wmma_f32_16x16x32_bf16(false, a0.v, false, b1.v, (short)0, acc[0][1], false, false);
    acc[1][0] = __builtin_amdgcn_wmma_f32_16x16x32_bf16(false, a1.v, false, b0.v, (short)0, acc[1][0], false, false);
    acc[1][1] = __builtin_amdgcn_wmma_f32_16x16x32_bf16(false, a1.v, false, b1.v, (short)0, acc[1][1], false, false);
    __syncthreads();
  }

  // epilogue: +bias, write bf16 twice: [b][co][hw] and transposed [b][hw][co]
  int hi = lane >> 4, l15 = lane & 15;
  #pragma unroll
  for (int tm = 0; tm < 2; ++tm) {
    #pragma unroll
    for (int tn = 0; tn < 2; ++tn) {
      #pragma unroll
      for (int v = 0; v < 8; ++v) {
        int co   = co_blk * 128 + wave_m * 32 + tm * 16 + v + 8 * hi;
        int wpos = w_base + wave_n * 32 + tn * 16 + l15;
        int hw   = h * W_ + wpos;
        float val = acc[tm][tn][v] + bc[co];
        img_bf [((size_t)bidx * C_  + co) * HW_ + hw] = (__bf16)val;
        imgT_bf[((size_t)bidx * HW_ + hw) * C_  + co] = (__bf16)val;
      }
    }
  }
}

// ---------------- adapter: class tokens (tiny) ----------------
__global__ void __launch_bounds__(256) class_tokens_kernel(const float* __restrict__ intra,
                                                           const float* __restrict__ inter,
                                                           float* __restrict__ ct) {
  int idx = blockIdx.x * 256 + threadIdx.x;
  if (idx >= B_ * NQ_ * NQ_) return;
  int p = idx % NQ_;
  int q = (idx / NQ_) % NQ_;
  int b = idx / (NQ_ * NQ_);
  const float* A  = intra + ((size_t)b * NQ_ + q) * C_;
  const float* Bp = inter + ((size_t)b * NQ_ + p) * C_;
  float s = 0.f;
  for (int c = 0; c < C_; ++c) s += A[c] * Bp[c];
  ct[idx] = s;
}

__global__ void __launch_bounds__(256) adapter_in_kernel(const float* __restrict__ intra,
                                                         const float* __restrict__ ct,
                                                         const float* __restrict__ Win,
                                                         const float* __restrict__ binv,
                                                         float* __restrict__ x1) {
  int bq = blockIdx.x;      // 1152
  int j  = threadIdx.x;     // 256
  const float* xa = intra + (size_t)bq * C_;
  const float* xb = ct + (size_t)bq * NQ_;
  float s = binv[j];
  for (int i = 0; i < C_; ++i)  s += xa[i] * Win[(size_t)i * C_ + j];
  for (int i = 0; i < NQ_; ++i) s += xb[i] * Win[(size_t)(C_ + i) * C_ + j];
  x1[(size_t)bq * C_ + j] = gelu_f(s);
}

__global__ void __launch_bounds__(256) adapter_wl_kernel(const float* __restrict__ x1,
                                                         const float* __restrict__ Wl,
                                                         const float* __restrict__ bl,
                                                         float* __restrict__ x2) {
  int bq = blockIdx.x;
  int j  = threadIdx.x;
  const float* xr = x1 + (size_t)bq * C_;
  float s = bl[j];
  for (int i = 0; i < C_; ++i) s += xr[i] * Wl[(size_t)i * C_ + j];
  x2[(size_t)bq * C_ + j] = s;
}

__global__ void __launch_bounds__(256) ln_kernel(const float* __restrict__ x,
                                                 const float* __restrict__ g,
                                                 const float* __restrict__ bb,
                                                 float* __restrict__ y) {
  int row = blockIdx.x;
  int j   = threadIdx.x;
  float v = x[(size_t)row * C_ + j];
  __shared__ float red[8];
  __shared__ float bval;
  int lane = j & 31, wid = j >> 5;
  float s = v;
  for (int o = 16; o; o >>= 1) s += __shfl_down(s, o, 32);
  if (lane == 0) red[wid] = s;
  __syncthreads();
  if (j == 0) { float t = 0.f; for (int i = 0; i < 8; ++i) t += red[i]; bval = t * (1.0f / C_); }
  __syncthreads();
  float mu = bval;
  float d  = v - mu;
  __syncthreads();
  float s2 = d * d;
  for (int o = 16; o; o >>= 1) s2 += __shfl_down(s2, o, 32);
  if (lane == 0) red[wid] = s2;
  __syncthreads();
  if (j == 0) { float t = 0.f; for (int i = 0; i < 8; ++i) t += red[i]; bval = t * (1.0f / C_); }
  __syncthreads();
  float var = bval;
  y[(size_t)row * C_ + j] = d * rsqrtf(var + 1e-5f) * g[j] + bb[j];
}

__global__ void __launch_bounds__(256) adapter_tok_kernel(const float* __restrict__ xn,
                                                          const float* __restrict__ Wa,
                                                          const float* __restrict__ ba,
                                                          float* __restrict__ y) {
  int idx = blockIdx.x * 256 + threadIdx.x;   // 1152*512
  int a  = idx & 63;
  int t  = (idx >> 6) & 7;
  int bq = idx >> 9;
  const float* xr = xn + (size_t)bq * C_;
  const float* w  = Wa + (size_t)t * (C_ * ADIM_) + a;
  float s = ba[t * ADIM_ + a];
  for (int c = 0; c < C_; ++c) s += xr[c] * w[(size_t)c * ADIM_];
  y[idx] = gelu_f(s);
}

__global__ void __launch_bounds__(256) mlp0_kernel(const float* __restrict__ y,
                                                   const float* __restrict__ W0,
                                                   const float* __restrict__ b0,
                                                   float* __restrict__ z0) {
  int bq = blockIdx.x;
  int j  = threadIdx.x;
  const float* r = y + (size_t)bq * HID_;
  float s = b0[j];
  for (int k = 0; k < HID_; ++k) s += r[k] * W0[(size_t)k * C_ + j];
  z0[(size_t)bq * C_ + j] = fmaxf(s, 0.f);
}

__global__ void __launch_bounds__(256) mlp1_kernel(const float* __restrict__ z0,
                                                   const float* __restrict__ W1,
                                                   const float* __restrict__ b1,
                                                   float* __restrict__ z1) {
  int bq = blockIdx.x;
  int j  = threadIdx.x;
  const float* r = z0 + (size_t)bq * C_;
  float s = b1[j];
  for (int k = 0; k < C_; ++k) s += r[k] * W1[(size_t)k * C_ + j];
  z1[(size_t)bq * C_ + j] = fmaxf(s, 0.f);
}

__global__ void __launch_bounds__(256) mlp2_kernel(const float* __restrict__ z1,
                                                   const float* __restrict__ W2,
                                                   const float* __restrict__ b2,
                                                   float* __restrict__ sparse) {
  int bq = blockIdx.x;
  int j  = threadIdx.x;
  const float* r = z1 + (size_t)bq * C_;
  float s = b2[j];
  for (int k = 0; k < C_; ++k) s += r[k] * W2[(size_t)k * C_ + j];
  sparse[(size_t)bq * C_ + j] = s;
}

__global__ void __launch_bounds__(256) pack_sparse_kernel(const float* __restrict__ sparse,
                                                          __bf16* __restrict__ sparse_bf) {
  int idx = blockIdx.x * 256 + threadIdx.x;   // 16*80*256
  if (idx >= B_ * NQP_ * C_) return;
  int c = idx & 255;
  int q = (idx >> 8) % NQP_;
  int b = idx / (NQP_ * C_);
  float v = (q < NQ_) ? sparse[((size_t)b * NQ_ + q) * C_ + c] : 0.0f;
  sparse_bf[idx] = (__bf16)v;
}

// ---------------- attention GEMM 1: S[b,q,d] = scale * sparse[q,:] . imgT[d,:] ----------------
// double-buffered, fully async staging (A and B are bf16 in memory)
__global__ void __launch_bounds__(256) attn_scores_kernel(const __bf16* __restrict__ sparse_bf,
                                                          const __bf16* __restrict__ imgT_bf,
                                                          float* __restrict__ S) {
  __shared__ __align__(16) __bf16 ldsA[2][NQP_ * 32];
  __shared__ __align__(16) __bf16 ldsB[2][256 * 32];
  int d_blk = blockIdx.x & 63;
  int bidx  = blockIdx.x >> 6;
  int d_base = d_blk * 256;
  int tid = threadIdx.x, lane = tid & 31, wid = tid >> 5;

  const f32x8 zero8 = {0.f,0.f,0.f,0.f,0.f,0.f,0.f,0.f};
  f32x8 acc[5][2];
  #pragma unroll
  for (int i = 0; i < 5; ++i) { acc[i][0] = zero8; acc[i][1] = zero8; }

  auto issue = [&](int ci0, int buf) {
    if (tid < 160) {
      int m = tid >> 1, half = tid & 1;
      cp32(sparse_bf + ((size_t)bidx * NQP_ + m) * C_ + ci0 + half * 16,
           &ldsA[buf][m * 32 + half * 16]);
    }
    cp64(imgT_bf + ((size_t)bidx * HW_ + d_base + tid) * C_ + ci0,
         &ldsB[buf][tid * 32]);
  };

  issue(0, 0);
  for (int kk = 0; kk < 8; ++kk) {
    int cur = kk & 1;
    if (kk < 7) issue((kk + 1) * 32, cur ^ 1);
#if USE_ASYNC
    if (kk < 7) { if (wid < 5) __builtin_amdgcn_s_wait_asynccnt(6);
                  else         __builtin_amdgcn_s_wait_asynccnt(4); }
    else        __builtin_amdgcn_s_wait_asynccnt(0);
#endif
    __syncthreads();
    FragBF b0, b1;
    b0.v = frag_b(&ldsB[cur][(wid * 32)      * 32]);
    b1.v = frag_b(&ldsB[cur][(wid * 32 + 16) * 32]);
    #pragma unroll
    for (int mt = 0; mt < 5; ++mt) {
      FragBF a; a.v = frag_a(&ldsA[cur][(mt * 16) * 32]);
      acc[mt][0] = __builtin_amdgcn_wmma_f32_16x16x32_bf16(false, a.v, false, b0.v, (short)0, acc[mt][0], false, false);
      acc[mt][1] = __builtin_amdgcn_wmma_f32_16x16x32_bf16(false, a.v, false, b1.v, (short)0, acc[mt][1], false, false);
    }
    __syncthreads();
  }

  const float scale = 0.0625f;   // 256^-0.5
  int hi = lane >> 4, l15 = lane & 15;
  #pragma unroll
  for (int mt = 0; mt < 5; ++mt)
    #pragma unroll
    for (int nt = 0; nt < 2; ++nt)
      #pragma unroll
      for (int v = 0; v < 8; ++v) {
        int q = mt * 16 + v + 8 * hi;
        int d = d_base + wid * 32 + nt * 16 + l15;
        S[((size_t)bidx * NQP_ + q) * HW_ + d] = acc[mt][nt][v] * scale;
      }
}

// ---------------- softmax + post-mask, write bf16 attn weights ----------------
__global__ void __launch_bounds__(256) softmax_mask_kernel(float* __restrict__ S,
                                                           const float* __restrict__ masks,
                                                           __bf16* __restrict__ attnm) {
  int q    = blockIdx.x % NQP_;
  int bidx = blockIdx.x / NQP_;
  int tid  = threadIdx.x;
  __bf16* arow = attnm + ((size_t)bidx * NQP_ + q) * HW_;
  if (q >= NQ_) {   // zero pad rows so GEMM2 sees defined data
    for (int i = tid; i < HW_; i += 256) arow[i] = (__bf16)0.0f;
    return;
  }
  float* srow = S + ((size_t)bidx * NQP_ + q) * HW_;
  const float* mrow = masks + ((size_t)bidx * NQ_ + q) * HW_;
  __shared__ float red[8];
  __shared__ float bval;
  int lane = tid & 31, wid = tid >> 5;

  float m = -1e30f;
  for (int i = tid; i < HW_; i += 256) m = fmaxf(m, srow[i]);
  for (int o = 16; o; o >>= 1) m = fmaxf(m, __shfl_down(m, o, 32));
  if (lane == 0) red[wid] = m;
  __syncthreads();
  if (tid == 0) { float t = red[0]; for (int i = 1; i < 8; ++i) t = fmaxf(t, red[i]); bval = t; }
  __syncthreads();
  float rmax = bval;
  __syncthreads();

  float s = 0.f;
  for (int i = tid; i < HW_; i += 256) { float e = expf(srow[i] - rmax); srow[i] = e; s += e; }
  for (int o = 16; o; o >>= 1) s += __shfl_down(s, o, 32);
  if (lane == 0) red[wid] = s;
  __syncthreads();
  if (tid == 0) { float t = 0.f; for (int i = 0; i < 8; ++i) t += red[i]; bval = t; }
  __syncthreads();
  float inv = 1.0f / bval;

  for (int i = tid; i < HW_; i += 256) arow[i] = (__bf16)(srow[i] * inv * mrow[i]);
}

__global__ void __launch_bounds__(256) zero_f32_kernel(float* __restrict__ p, int n) {
  int idx = blockIdx.x * 256 + threadIdx.x;
  if (idx < n) p[idx] = 0.0f;
}

// ---------------- attention GEMM 2: attn2[b,q,c] += sum_d attnm[q,d]*img[c,d] ----------------
// split-K: 8 segments of 2048, native f32 atomics; double-buffered async staging
__global__ void __launch_bounds__(256) attn_out_kernel(const __bf16* __restrict__ attnm,
                                                       const __bf16* __restrict__ img_bf,
                                                       float* __restrict__ attn2) {
  __shared__ __align__(16) __bf16 ldsA[2][NQP_ * 32];
  __shared__ __align__(16) __bf16 ldsB[2][256 * 32];
  int kseg = blockIdx.x & 7;
  int bidx = blockIdx.x >> 3;
  int tid = threadIdx.x, lane = tid & 31, wid = tid >> 5;

  const f32x8 zero8 = {0.f,0.f,0.f,0.f,0.f,0.f,0.f,0.f};
  f32x8 acc[5][2];
  #pragma unroll
  for (int i = 0; i < 5; ++i) { acc[i][0] = zero8; acc[i][1] = zero8; }

  int dbase = kseg * 2048;
  auto issue = [&](int dc, int buf) {
    int d0 = dbase + dc;
    if (tid < 160) {
      int m = tid >> 1, half = tid & 1;
      cp32(attnm + ((size_t)bidx * NQP_ + m) * HW_ + d0 + half * 16,
           &ldsA[buf][m * 32 + half * 16]);
    }
    cp64(img_bf + ((size_t)bidx * C_ + tid) * HW_ + d0,
         &ldsB[buf][tid * 32]);
  };

  issue(0, 0);
  for (int kk = 0; kk < 64; ++kk) {
    int cur = kk & 1;
    if (kk < 63) issue((kk + 1) * 32, cur ^ 1);
#if USE_ASYNC
    if (kk < 63) { if (wid < 5) __builtin_amdgcn_s_wait_asynccnt(6);
                   else         __builtin_amdgcn_s_wait_asynccnt(4); }
    else         __builtin_amdgcn_s_wait_asynccnt(0);
#endif
    __syncthreads();
    FragBF b0, b1;
    b0.v = frag_b(&ldsB[cur][(wid * 32)      * 32]);
    b1.v = frag_b(&ldsB[cur][(wid * 32 + 16) * 32]);
    #pragma unroll
    for (int mt = 0; mt < 5; ++mt) {
      FragBF a; a.v = frag_a(&ldsA[cur][(mt * 16) * 32]);
      acc[mt][0] = __builtin_amdgcn_wmma_f32_16x16x32_bf16(false, a.v, false, b0.v, (short)0, acc[mt][0], false, false);
      acc[mt][1] = __builtin_amdgcn_wmma_f32_16x16x32_bf16(false, a.v, false, b1.v, (short)0, acc[mt][1], false, false);
    }
    __syncthreads();
  }

  int hi = lane >> 4, l15 = lane & 15;
  #pragma unroll
  for (int mt = 0; mt < 5; ++mt)
    #pragma unroll
    for (int nt = 0; nt < 2; ++nt)
      #pragma unroll
      for (int v = 0; v < 8; ++v) {
        int q = mt * 16 + v + 8 * hi;
        int c = wid * 32 + nt * 16 + l15;
        (void)__hip_atomic_fetch_add(&attn2[((size_t)bidx * NQP_ + q) * C_ + c],
                                     acc[mt][nt][v],
                                     __ATOMIC_RELAXED, __HIP_MEMORY_SCOPE_AGENT);
      }
}

// ---------------- epilogue ----------------
__global__ void __launch_bounds__(256) residual_kernel(const float* __restrict__ attn2,
                                                       const float* __restrict__ sparse,
                                                       const float* __restrict__ alpha,
                                                       float* __restrict__ tbuf) {
  int bq = blockIdx.x;       // 1152
  int c  = threadIdx.x;
  int b = bq / NQ_, q = bq % NQ_;
  float a2 = attn2[((size_t)b * NQP_ + q) * C_ + c];
  tbuf[(size_t)bq * C_ + c] = a2 + sparse[(size_t)bq * C_ + c] * alpha[c];
}

__global__ void __launch_bounds__(256) out_h_kernel(const float* __restrict__ tbuf,
                                                    const float* __restrict__ Wo1,
                                                    const float* __restrict__ bo1,
                                                    float* __restrict__ hbuf) {
  int idx = blockIdx.x * 256 + threadIdx.x;   // 1152*128
  int j  = idx & 127;
  int bq = idx >> 7;
  const float* r = tbuf + (size_t)bq * C_;
  float s = bo1[j];
  for (int c = 0; c < C_; ++c) s += r[c] * Wo1[(size_t)c * 128 + j];
  hbuf[idx] = gelu_f(s);
}

__global__ void __launch_bounds__(256) out_final_kernel(const float* __restrict__ hbuf,
                                                        const float* __restrict__ Wo2,
                                                        const float* __restrict__ bo2,
                                                        float* __restrict__ out) {
  int bq = blockIdx.x;       // 1152
  int c  = threadIdx.x;
  const float* r = hbuf + (size_t)bq * 128;
  float s = bo2[c];
  for (int j = 0; j < 128; ++j) s += r[j] * Wo2[(size_t)j * C_ + c];
  out[(size_t)bq * C_ + c] = s;
}

// ================= host launch =================
extern "C" void kernel_launch(void* const* d_in, const int* in_sizes, int n_in,
                              void* d_out, int out_size, void* d_ws, size_t ws_size,
                              hipStream_t stream) {
  const float* image_embed = (const float*)d_in[0];
  const float* inter  = (const float*)d_in[1];
  const float* intra  = (const float*)d_in[2];
  const float* masks  = (const float*)d_in[3];
  const float* Wc     = (const float*)d_in[4];
  const float* bc     = (const float*)d_in[5];
  const float* Win    = (const float*)d_in[6];
  const float* binv   = (const float*)d_in[7];
  const float* Wl     = (const float*)d_in[8];
  const float* bl     = (const float*)d_in[9];
  const float* ln_g   = (const float*)d_in[10];
  const float* ln_b   = (const float*)d_in[11];
  const float* Wa     = (const float*)d_in[12];
  const float* ba     = (const float*)d_in[13];
  const float* W0     = (const float*)d_in[14];
  const float* b0     = (const float*)d_in[15];
  const float* W1     = (const float*)d_in[16];
  const float* b1     = (const float*)d_in[17];
  const float* W2     = (const float*)d_in[18];
  const float* b2     = (const float*)d_in[19];
  const float* Wo1    = (const float*)d_in[20];
  const float* bo1    = (const float*)d_in[21];
  const float* Wo2    = (const float*)d_in[22];
  const float* bo2    = (const float*)d_in[23];
  const float* alpha  = (const float*)d_in[24];
  float* out = (float*)d_out;

  char* wp = (char*)d_ws;
  auto carve = [&](size_t bytes) -> void* {
    void* r = (void*)wp;
    wp += (bytes + 255) & ~((size_t)255);
    return r;
  };
  __bf16* Wp        = (__bf16*)carve((size_t)9 * C_ * C_ * 2);
  __bf16* img_bf    = (__bf16*)carve((size_t)B_ * C_ * HW_ * 2);
  __bf16* imgT_bf   = (__bf16*)carve((size_t)B_ * HW_ * C_ * 2);
  float*  S         = (float*) carve((size_t)B_ * NQP_ * HW_ * 4);
  __bf16* attnm     = (__bf16*)carve((size_t)B_ * NQP_ * HW_ * 2);
  float*  ct        = (float*) carve((size_t)B_ * NQ_ * NQ_ * 4);
  float*  x1        = (float*) carve((size_t)B_ * NQ_ * C_ * 4);
  float*  x2        = (float*) carve((size_t)B_ * NQ_ * C_ * 4);
  float*  xn        = (float*) carve((size_t)B_ * NQ_ * C_ * 4);
  float*  ybuf      = (float*) carve((size_t)B_ * NQ_ * HID_ * 4);
  float*  z0        = (float*) carve((size_t)B_ * NQ_ * C_ * 4);
  float*  z1        = (float*) carve((size_t)B_ * NQ_ * C_ * 4);
  float*  sparse    = (float*) carve((size_t)B_ * NQ_ * C_ * 4);
  __bf16* sparse_bf = (__bf16*)carve((size_t)B_ * NQP_ * C_ * 2);
  float*  attn2     = (float*) carve((size_t)B_ * NQP_ * C_ * 4);
  float*  tbuf      = (float*) carve((size_t)B_ * NQ_ * C_ * 4);
  float*  hbuf      = (float*) carve((size_t)B_ * NQ_ * 128 * 4);

  const int BQ = B_ * NQ_;  // 1152

  pack_weights_kernel<<<(9 * C_ * C_ + 255) / 256, 256, 0, stream>>>(Wc, Wp);
  conv_wmma_kernel<<<B_ * H_ * 2 * 2, 256, 0, stream>>>(image_embed, Wp, bc, img_bf, imgT_bf);

  class_tokens_kernel<<<(B_ * NQ_ * NQ_ + 255) / 256, 256, 0, stream>>>(intra, inter, ct);
  adapter_in_kernel<<<BQ, 256, 0, stream>>>(intra, ct, Win, binv, x1);
  adapter_wl_kernel<<<BQ, 256, 0, stream>>>(x1, Wl, bl, x2);
  ln_kernel<<<BQ, 256, 0, stream>>>(x2, ln_g, ln_b, xn);
  adapter_tok_kernel<<<BQ * HID_ / 256, 256, 0, stream>>>(xn, Wa, ba, ybuf);
  mlp0_kernel<<<BQ, 256, 0, stream>>>(ybuf, W0, b0, z0);
  mlp1_kernel<<<BQ, 256, 0, stream>>>(z0, W1, b1, z1);
  mlp2_kernel<<<BQ, 256, 0, stream>>>(z1, W2, b2, sparse);
  pack_sparse_kernel<<<(B_ * NQP_ * C_ + 255) / 256, 256, 0, stream>>>(sparse, sparse_bf);

  attn_scores_kernel<<<B_ * (HW_ / 256), 256, 0, stream>>>(sparse_bf, imgT_bf, S);
  softmax_mask_kernel<<<B_ * NQP_, 256, 0, stream>>>(S, masks, attnm);
  zero_f32_kernel<<<(B_ * NQP_ * C_ + 255) / 256, 256, 0, stream>>>(attn2, B_ * NQP_ * C_);
  attn_out_kernel<<<B_ * 8, 256, 0, stream>>>(attnm, img_bf, attn2);

  residual_kernel<<<BQ, 256, 0, stream>>>(attn2, sparse, alpha, tbuf);
  out_h_kernel<<<BQ * 128 / 256, 256, 0, stream>>>(tbuf, Wo1, bo1, hbuf);
  out_final_kernel<<<BQ, 256, 0, stream>>>(hbuf, Wo2, bo2, out);
}
